// SphericalFourierNeuralOperatorNet_25933012533835
// MI455X (gfx1250) — matrix operand order repaired and checked by reading, not compiled, not run on hardware
//
#include <hip/hip_runtime.h>
#include <hip/hip_bf16.h>
#include <math.h>

// ---------------- problem constants (match reference) ----------------
#define BATCH   2
#define CIN     8
#define COUT    8
#define NLAT    128
#define NLON    256
#define EMBED   256
#define NLAYERS 4
#define HIDDEN  512
#define LMAX    128
#define MMAX    129          // NLON/2 + 1
#define PIX     (NLAT * NLON)            // 32768
#define HTOT    (BATCH * EMBED * PIX)    // 16,777,216 elements
#define BC      (BATCH * EMBED)          // 512
#define BM      (BATCH * MMAX)           // 258
#define FE      (MMAX * BC * LMAX)       // 8,454,144  (== 65536*129 == 128*258*256)
#define K2PAD   288                      // 2*MMAX=258 padded to mult of 32

typedef __attribute__((ext_vector_type(16))) _Float16 v16h;
typedef __attribute__((ext_vector_type(8)))  float    v8f;

__device__ __forceinline__ float gelu_f(float x) {
  // tanh approximation (jax.nn.gelu default)
  float x3 = x * x * x;
  return 0.5f * x * (1.0f + tanhf(0.7978845608028654f * (x + 0.044715f * x3)));
}

// ================= WMMA GEMM: C[M,N] = act(alpha*A(M,Kpad)·Bt(N,Kpad)^T + bias [+C]) ========
// A row-major [M,Kpad] f16, Bt row-major [N,Kpad] f16 (i.e. B^T), Kpad % 32 == 0 (>=128 here),
// pad = zeros. Block tile 128x128, 8 waves (wave32) arranged 2(M) x 4(N); wave tile 64x32 =
// 4x2 WMMA 16x16x32 fragments (8 v_wmma per K-tile per wave, 12 ds_load_b128 -> 1.5 ds/wmma).
// Tiles staged to LDS with GLOBAL_LOAD_ASYNC_TO_LDS_B128 (ASYNCcnt), double-buffered,
// pipelined one K-tile ahead; last two iterations peeled so the steady-state loop has
// branch-free s_wait_asynccnt 0x4 + issue.
// Load-side bounds checks are intentionally absent: operand buffers are workspace-owned,
// K-pad regions are zero-filled by the pack kernels, and out-of-range M/N tile rows only
// produce garbage accumulators that the bounds-checked epilogue never stores.

#define TM 128
#define TN 128
#define TK 32
#define LDT (TK + 8)   // LDS row stride in halves (80B, 16B-aligned rows)

// A fragment (16x32 f16): lane r=lane&15 row; half h selects K block of 16 (k = 16h + s).
__device__ __forceinline__ v16h frag_a(const _Float16* base, int lane) {
  int r = lane & 15, h = lane >> 4;
  const _Float16* p = base + r * LDT;
  union { uint4 q[2]; v16h v; } u;
  u.q[0] = *(const uint4*)(p + 16 * h);       // elems 0..7  <- K = 16h .. 16h+7
  u.q[1] = *(const uint4*)(p + 16 * h + 8);   // elems 8..15 <- K = 16h+8 ..
  return u.v;
}
// B fragment (32x16 f16): lane n=lane&15 column; half kh selects K block of 16.
__device__ __forceinline__ v16h frag_b(const _Float16* base, int lane) {
  int n = lane & 15, kh = lane >> 4;
  const _Float16* p = base + n * LDT + 16 * kh;
  union { uint4 q[2]; v16h v; } u;
  u.q[0] = *(const uint4*)(p);
  u.q[1] = *(const uint4*)(p + 8);
  return u.v;
}

__global__ __launch_bounds__(256)
void gemm_f16_wmma(const _Float16* __restrict__ A, const _Float16* __restrict__ Bt,
                   float* __restrict__ Cf, _Float16* __restrict__ Ch,
                   int M, int N, int Kpad,
                   long long sA, long long sB, long long sC,
                   const float* __restrict__ bias,
                   float alpha, int addC, int act, int out_half) {
  __shared__ __align__(16) _Float16 As[2][TM][LDT];
  __shared__ __align__(16) _Float16 Bs[2][TN][LDT];

  const _Float16* Ap = A  + (long long)blockIdx.z * sA;
  const _Float16* Bp = Bt + (long long)blockIdx.z * sB;
  const long long  cb = (long long)blockIdx.z * sC;

  const int m0 = blockIdx.y * TM;
  const int n0 = blockIdx.x * TN;
  const int tid  = threadIdx.x;
  const int lane = tid & 31;
  const int wave = tid >> 5;
  const int wm = wave >> 2;   // 0..1  (M direction, 64 rows each)
  const int wn = wave & 3;    // 0..3  (N direction, 32 cols each)

  // per-thread tile-copy coordinates: A and B tiles are each 512 uint4 (2/thread).
  const int rT = tid >> 2;        // rows rT and rT+64
  const int cq = tid & 3;         // 16B chunk within the 32-half K-slice

  const int ktiles = Kpad / TK;   // >= 4 for every GEMM in this network

  // async-copy one K tile (4 instructions per wave -> ASYNCcnt += 4)
  auto issue = [&](int j) {
    const int k0 = j * TK;
    const int buf = j & 1;
    {
      const _Float16* gp = Ap + (long long)(m0 + rT) * Kpad + k0 + cq * 8;
      unsigned lp = (unsigned)(uintptr_t)&As[buf][rT][cq * 8];
      asm volatile("global_load_async_to_lds_b128 %0, %1, off" :: "v"(lp), "v"(gp) : "memory");
    }
    {
      const _Float16* gp = Ap + (long long)(m0 + rT + 64) * Kpad + k0 + cq * 8;
      unsigned lp = (unsigned)(uintptr_t)&As[buf][rT + 64][cq * 8];
      asm volatile("global_load_async_to_lds_b128 %0, %1, off" :: "v"(lp), "v"(gp) : "memory");
    }
    {
      const _Float16* gp = Bp + (long long)(n0 + rT) * Kpad + k0 + cq * 8;
      unsigned lp = (unsigned)(uintptr_t)&Bs[buf][rT][cq * 8];
      asm volatile("global_load_async_to_lds_b128 %0, %1, off" :: "v"(lp), "v"(gp) : "memory");
    }
    {
      const _Float16* gp = Bp + (long long)(n0 + rT + 64) * Kpad + k0 + cq * 8;
      unsigned lp = (unsigned)(uintptr_t)&Bs[buf][rT + 64][cq * 8];
      asm volatile("global_load_async_to_lds_b128 %0, %1, off" :: "v"(lp), "v"(gp) : "memory");
    }
  };

  v8f acc[4][2] = {};

  auto compute = [&](int j) {
    const int buf = j & 1;
    v16h a0 = frag_a(&As[buf][wm * 64 + 0][0], lane);
    v16h a1 = frag_a(&As[buf][wm * 64 + 16][0], lane);
    v16h a2 = frag_a(&As[buf][wm * 64 + 32][0], lane);
    v16h a3 = frag_a(&As[buf][wm * 64 + 48][0], lane);
    v16h b0 = frag_b(&Bs[buf][wn * 32 + 0][0], lane);
    v16h b1 = frag_b(&Bs[buf][wn * 32 + 16][0], lane);
    acc[0][0] = __builtin_amdgcn_wmma_f32_16x16x32_f16(false, a0, false, b0, (short)0, acc[0][0], false, false);
    acc[0][1] = __builtin_amdgcn_wmma_f32_16x16x32_f16(false, a0, false, b1, (short)0, acc[0][1], false, false);
    acc[1][0] = __builtin_amdgcn_wmma_f32_16x16x32_f16(false, a1, false, b0, (short)0, acc[1][0], false, false);
    acc[1][1] = __builtin_amdgcn_wmma_f32_16x16x32_f16(false, a1, false, b1, (short)0, acc[1][1], false, false);
    acc[2][0] = __builtin_amdgcn_wmma_f32_16x16x32_f16(false, a2, false, b0, (short)0, acc[2][0], false, false);
    acc[2][1] = __builtin_amdgcn_wmma_f32_16x16x32_f16(false, a2, false, b1, (short)0, acc[2][1], false, false);
    acc[3][0] = __builtin_amdgcn_wmma_f32_16x16x32_f16(false, a3, false, b0, (short)0, acc[3][0], false, false);
    acc[3][1] = __builtin_amdgcn_wmma_f32_16x16x32_f16(false, a3, false, b1, (short)0, acc[3][1], false, false);
  };

  issue(0);
  issue(1);

  int j = 0;
  // steady state: branch-free wait (tile j done, tile j+1 in flight) + unconditional issue
  for (; j + 2 < ktiles; ++j) {
    asm volatile("s_wait_asynccnt 0x4" ::: "memory");
    __syncthreads();
    compute(j);
    __syncthreads();                 // all waves done reading buf before overwrite
    issue(j + 2);
  }
  // peeled: second-to-last tile (tile j+1 still in flight)
  asm volatile("s_wait_asynccnt 0x4" ::: "memory");
  __syncthreads();
  compute(j);
  __syncthreads();
  ++j;
  // peeled: last tile (drain)
  asm volatile("s_wait_asynccnt 0x0" ::: "memory");
  __syncthreads();
  compute(j);

  // epilogue: C/D layout -> lane n = lane&15, row = g + 8*(lane>>4) per fragment
  const int nl = lane & 15, hh = lane >> 4;
  #pragma unroll
  for (int i = 0; i < 4; ++i) {
    #pragma unroll
    for (int jj = 0; jj < 2; ++jj) {
      int col = n0 + wn * 32 + jj * 16 + nl;
      if (col >= N) continue;
      float bv = bias ? bias[col] : 0.0f;
      int rbase = m0 + wm * 64 + i * 16 + hh * 8;
      #pragma unroll
      for (int g = 0; g < 8; ++g) {
        int r = rbase + g;
        if (r >= M) continue;
        long long off = cb + (long long)r * N + col;
        float v = alpha * acc[i][jj][g] + bv;
        if (addC)  v += Cf[off];
        if (act)   v = gelu_f(v);
        if (out_half) Ch[off] = (_Float16)v;
        else          Cf[off] = v;
      }
    }
  }
}

// ================= elementwise / pack kernels =================

__global__ void k_cvt_h(const float* __restrict__ in, _Float16* __restrict__ out, long long n) {
  long long t = (long long)blockIdx.x * blockDim.x + threadIdx.x;
  if (t < n) out[t] = (_Float16)in[t];
}

__global__ void k_encoder(const float* __restrict__ x, const float* __restrict__ w,
                          const float* __restrict__ b, const float* __restrict__ pos,
                          float* __restrict__ h) {
  long long t = (long long)blockIdx.x * blockDim.x + threadIdx.x;
  if (t >= (long long)HTOT) return;
  int bb = (int)(t / ((long long)EMBED * PIX));
  int r  = (int)(t % ((long long)EMBED * PIX));
  int o  = r / PIX, p = r % PIX;
  float s = 0.0f;
  #pragma unroll
  for (int c = 0; c < CIN; ++c)
    s += x[((long long)bb * CIN + c) * PIX + p] * w[o * CIN + c];
  h[t] = s + b[o] + pos[(long long)o * PIX + p];
}

__global__ void k_rfft_mat(_Float16* __restrict__ Fre, _Float16* __restrict__ Fim) {
  int t = blockIdx.x * blockDim.x + threadIdx.x;
  if (t >= MMAX * NLON) return;
  int m = t / NLON, w = t % NLON;
  float th = 6.283185307179586f * (float)(w * m) / (float)NLON;
  float sc = 6.283185307179586f / (float)NLON;
  Fre[t] = (_Float16)(sc * cosf(th));
  Fim[t] = (_Float16)(-sc * sinf(th));
}

__global__ void k_irfft_mat(_Float16* __restrict__ G) {
  int t = blockIdx.x * blockDim.x + threadIdx.x;
  if (t >= NLON * K2PAD) return;
  int w = t / K2PAD, c = t % K2PAD;
  float v = 0.0f;
  if (c < MMAX) {
    float eps = (c == 0 || c == MMAX - 1) ? 1.0f : 2.0f;
    v = eps * cosf(6.283185307179586f * (float)(w * c) / (float)NLON);
  } else if (c < 2 * MMAX) {
    int m = c - MMAX;
    float eps = (m == 0 || m == MMAX - 1) ? 1.0f : 2.0f;
    v = -eps * sinf(6.283185307179586f * (float)(w * m) / (float)NLON);
  }
  G[t] = (_Float16)v;
}

// iwt_t[m][k][l] = isht_wt[m][l][k]
__global__ void k_cvt_isht(const float* __restrict__ w, _Float16* __restrict__ o) {
  int t = blockIdx.x * blockDim.x + threadIdx.x;
  if (t >= MMAX * NLAT * LMAX) return;
  int m = t / (NLAT * LMAX), r = t % (NLAT * LMAX);
  int k = r / LMAX, l = r % LMAX;
  o[t] = (_Float16)w[((long long)m * LMAX + l) * NLAT + k];
}

// A_sht[m][bc][k] = Xf[(bc*NLAT + k)*MMAX + m]
__global__ void k_pack_sht(const float* __restrict__ Xf, _Float16* __restrict__ A) {
  long long t = (long long)blockIdx.x * blockDim.x + threadIdx.x;
  if (t >= (long long)FE) return;
  int m = (int)(t / (BC * NLAT));
  int r = (int)(t % (BC * NLAT));
  int bc = r / NLAT, k = r % NLAT;
  A[t] = (_Float16)Xf[((long long)bc * NLAT + k) * MMAX + m];
}

// Adh[l][bm][i] = Hs[m][b*EMBED+i][l]   (bm = b*MMAX + m)
__global__ void k_pack_dh(const float* __restrict__ Hs, _Float16* __restrict__ A) {
  long long t = (long long)blockIdx.x * blockDim.x + threadIdx.x;
  if (t >= (long long)LMAX * BM * EMBED) return;
  int l = (int)(t / (BM * EMBED));
  int r = (int)(t % (BM * EMBED));
  int bm = r / EMBED, i = r % EMBED;
  int b = bm / MMAX, m = bm % MMAX;
  A[t] = (_Float16)Hs[((long long)m * BC + b * EMBED + i) * LMAX + l];
}

// A2[m][b*EMBED+o][l] = Cdh[l][b*MMAX+m][o]
__global__ void k_pack_isht(const float* __restrict__ C, _Float16* __restrict__ A) {
  long long t = (long long)blockIdx.x * blockDim.x + threadIdx.x;
  if (t >= (long long)FE) return;
  int m = (int)(t / (BC * LMAX));
  int r = (int)(t % (BC * LMAX));
  int bo = r / LMAX, l = r % LMAX;
  int b = bo / EMBED, o = bo % EMBED;
  A[t] = (_Float16)C[((long long)l * BM + b * MMAX + m) * EMBED + o];
}

// A3[(bc*NLAT+k)][c] : c<129 -> Yre[m=c][bc][k]; 129<=c<258 -> Yim; else 0
__global__ void k_pack_idft(const float* __restrict__ Yre, const float* __restrict__ Yim,
                            _Float16* __restrict__ A) {
  long long t = (long long)blockIdx.x * blockDim.x + threadIdx.x;
  if (t >= (long long)BC * NLAT * K2PAD) return;
  int row = (int)(t / K2PAD), c = (int)(t % K2PAD);
  int bc = row / NLAT, k = row % NLAT;
  float v = 0.0f;
  if (c < MMAX)          v = Yre[((long long)c * BC + bc) * NLAT + k];
  else if (c < 2 * MMAX) v = Yim[((long long)(c - MMAX) * BC + bc) * NLAT + k];
  A[t] = (_Float16)v;
}

__global__ void k_gelu_add(const float* __restrict__ g, const float* __restrict__ res,
                           float* __restrict__ h, long long n) {
  long long t = (long long)blockIdx.x * blockDim.x + threadIdx.x;
  if (t < n) h[t] = gelu_f(g[t]) + res[t];
}

// hT[(b*PIX+p)][c] = h[b][c][p]
__global__ void k_pack_hT(const float* __restrict__ h, _Float16* __restrict__ o) {
  long long t = (long long)blockIdx.x * blockDim.x + threadIdx.x;
  if (t >= (long long)HTOT) return;
  int row = (int)(t / EMBED), c = (int)(t % EMBED);
  int b = row / PIX, p = row % PIX;
  o[t] = (_Float16)h[((long long)b * EMBED + c) * PIX + p];
}

// h[b][c][p] += C2[(b*PIX+p)][c]
__global__ void k_mlp_res(const float* __restrict__ C2, float* __restrict__ h) {
  long long t = (long long)blockIdx.x * blockDim.x + threadIdx.x;
  if (t >= (long long)HTOT) return;
  int b = (int)(t / ((long long)EMBED * PIX));
  int r = (int)(t % ((long long)EMBED * PIX));
  int c = r / PIX, p = r % PIX;
  h[t] += C2[((long long)b * PIX + p) * EMBED + c];
}

// wre[l][o][i] / wim[l][o][i] from spec_w[layer][i][o][l][{0,1}]
__global__ void k_pack_spec(const float* __restrict__ spec, _Float16* __restrict__ wre,
                            _Float16* __restrict__ wim) {
  long long t = (long long)blockIdx.x * blockDim.x + threadIdx.x;
  if (t >= (long long)LMAX * EMBED * EMBED) return;
  int l = (int)(t / (EMBED * EMBED));
  int r = (int)(t % (EMBED * EMBED));
  int o = r / EMBED, i = r % EMBED;
  long long idx = (((long long)i * EMBED + o) * LMAX + l) * 2;
  wre[t] = (_Float16)spec[idx];
  wim[t] = (_Float16)spec[idx + 1];
}

__global__ void k_decoder(const float* __restrict__ h, const float* __restrict__ w,
                          const float* __restrict__ b, float* __restrict__ out) {
  long long t = (long long)blockIdx.x * blockDim.x + threadIdx.x;
  if (t >= (long long)BATCH * COUT * PIX) return;
  int bb = (int)(t / ((long long)COUT * PIX));
  int r  = (int)(t % ((long long)COUT * PIX));
  int o  = r / PIX, p = r % PIX;
  float s = 0.0f;
  for (int c = 0; c < EMBED; ++c)
    s += h[((long long)bb * EMBED + c) * PIX + p] * w[o * EMBED + c];
  out[t] = s + b[o];
}

// ================= host orchestration =================

static inline void gemm(hipStream_t s, const _Float16* A, const _Float16* Bt,
                        float* Cf, _Float16* Ch, int M, int N, int Kpad, int batch,
                        long long sA, long long sB, long long sC,
                        const float* bias, float alpha, int addC, int act, int outh) {
  dim3 grid((N + TN - 1) / TN, (M + TM - 1) / TM, batch);
  gemm_f16_wmma<<<grid, dim3(256), 0, s>>>(A, Bt, Cf, Ch, M, N, Kpad, sA, sB, sC,
                                           bias, alpha, addC, act, outh);
}

#define L1D(kern, n, ...) kern<<<(unsigned)(((n) + 255) / 256), 256, 0, stream>>>(__VA_ARGS__)

extern "C" void kernel_launch(void* const* d_in, const int* in_sizes, int n_in,
                              void* d_out, int out_size, void* d_ws, size_t ws_size,
                              hipStream_t stream) {
  const float* x        = (const float*)d_in[0];
  const float* w_enc    = (const float*)d_in[1];
  const float* b_enc    = (const float*)d_in[2];
  const float* pos      = (const float*)d_in[3];
  const float* spec_w   = (const float*)d_in[4];
  const float* w1       = (const float*)d_in[5];
  const float* b1       = (const float*)d_in[6];
  const float* w2       = (const float*)d_in[7];
  const float* b2       = (const float*)d_in[8];
  const float* w_dec    = (const float*)d_in[9];
  const float* b_dec    = (const float*)d_in[10];
  const float* sht_wt   = (const float*)d_in[11];
  const float* isht_wt  = (const float*)d_in[12];
  float* out = (float*)d_out;

  // ---- workspace bump allocator (256B aligned) ----
  uint8_t* base = (uint8_t*)d_ws;
  size_t cur = 0;
  auto alloc = [&](size_t bytes) -> uint8_t* {
    uint8_t* p = base + cur;
    cur += (bytes + 255) & ~(size_t)255;
    return p;
  };
  float*     h     = (float*)alloc((size_t)HTOT * 4);
  float*     hres  = (float*)alloc((size_t)HTOT * 4);
  float*     g1    = (float*)alloc((size_t)HTOT * 4);
  _Float16*  abufA = (_Float16*)alloc((size_t)65536 * 512 * 2);
  _Float16*  abufB = (_Float16*)alloc((size_t)65536 * 512 * 2);
  float*     f1    = (float*)alloc((size_t)FE * 4);
  float*     f2    = (float*)alloc((size_t)FE * 4);
  float*     f3    = (float*)alloc((size_t)FE * 4);
  float*     f4    = (float*)alloc((size_t)FE * 4);
  _Float16*  sA1   = (_Float16*)alloc((size_t)FE * 2);
  _Float16*  sA2   = (_Float16*)alloc((size_t)FE * 2);
  _Float16*  wb1   = (_Float16*)alloc((size_t)LMAX * EMBED * EMBED * 2);
  _Float16*  wb2   = (_Float16*)alloc((size_t)LMAX * EMBED * EMBED * 2);
  _Float16*  swt   = (_Float16*)alloc((size_t)MMAX * LMAX * NLAT * 2);
  _Float16*  iwt   = (_Float16*)alloc((size_t)MMAX * NLAT * LMAX * 2);
  _Float16*  Fre   = (_Float16*)alloc((size_t)MMAX * NLON * 2);
  _Float16*  Fim   = (_Float16*)alloc((size_t)MMAX * NLON * 2);
  _Float16*  Gm    = (_Float16*)alloc((size_t)NLON * K2PAD * 2);
  _Float16*  w1h   = (_Float16*)alloc((size_t)HIDDEN * EMBED * 2);
  _Float16*  w2h   = (_Float16*)alloc((size_t)EMBED * HIDDEN * 2);
  (void)alloc((size_t)1 << 20);   // tail guard: unbounded tile loads may over-read

  // ---- constants ----
  L1D(k_rfft_mat,  MMAX * NLON, Fre, Fim);
  L1D(k_irfft_mat, NLON * K2PAD, Gm);
  L1D(k_cvt_h, (long long)MMAX * LMAX * NLAT, sht_wt, swt, (long long)MMAX * LMAX * NLAT);
  L1D(k_cvt_isht, MMAX * NLAT * LMAX, isht_wt, iwt);

  // ---- encoder ----
  L1D(k_encoder, HTOT, x, w_enc, b_enc, pos, h);

  for (int l = 0; l < NLAYERS; ++l) {
    // residual save
    hipMemcpyAsync(hres, h, (size_t)HTOT * 4, hipMemcpyDeviceToDevice, stream);

    // (1) rFFT as GEMM: [65536 x 256] x [129 x 256]^T  (re & im)
    L1D(k_cvt_h, (long long)HTOT, h, abufA, (long long)HTOT);
    gemm(stream, abufA, Fre, f1, nullptr, BC * NLAT, MMAX, NLON, 1, 0, 0, 0, nullptr, 1.f, 0, 0, 0);
    gemm(stream, abufA, Fim, f2, nullptr, BC * NLAT, MMAX, NLON, 1, 0, 0, 0, nullptr, 1.f, 0, 0, 0);

    // (2) Legendre SHT, batched over m: [512 x 128] x [128 x 128]^T
    L1D(k_pack_sht, (long long)FE, f1, sA1);
    L1D(k_pack_sht, (long long)FE, f2, sA2);
    gemm(stream, sA1, swt, f3, nullptr, BC, LMAX, NLAT, MMAX,
         (long long)BC * NLAT, (long long)LMAX * NLAT, (long long)BC * LMAX, nullptr, 1.f, 0, 0, 0);
    gemm(stream, sA2, swt, f4, nullptr, BC, LMAX, NLAT, MMAX,
         (long long)BC * NLAT, (long long)LMAX * NLAT, (long long)BC * LMAX, nullptr, 1.f, 0, 0, 0);

    // (3) dhconv (complex), batched over l: [258 x 256] x [256 x 256]^T, 4 real GEMMs
    L1D(k_pack_dh, (long long)LMAX * BM * EMBED, f3, sA1);   // re
    L1D(k_pack_dh, (long long)LMAX * BM * EMBED, f4, sA2);   // im
    L1D(k_pack_spec, (long long)LMAX * EMBED * EMBED,
        spec_w + (long long)l * EMBED * EMBED * LMAX * 2, wb1, wb2);
    long long sAd = (long long)BM * EMBED, sBd = (long long)EMBED * EMBED, sCd = (long long)BM * EMBED;
    gemm(stream, sA1, wb1, f1, nullptr, BM, EMBED, EMBED, LMAX, sAd, sBd, sCd, nullptr,  1.f, 0, 0, 0); // re = hr*wr
    gemm(stream, sA2, wb2, f1, nullptr, BM, EMBED, EMBED, LMAX, sAd, sBd, sCd, nullptr, -1.f, 1, 0, 0); //    - hi*wi
    gemm(stream, sA1, wb2, f2, nullptr, BM, EMBED, EMBED, LMAX, sAd, sBd, sCd, nullptr,  1.f, 0, 0, 0); // im = hr*wi
    gemm(stream, sA2, wb1, f2, nullptr, BM, EMBED, EMBED, LMAX, sAd, sBd, sCd, nullptr,  1.f, 1, 0, 0); //    + hi*wr

    // (4) inverse Legendre, batched over m
    L1D(k_pack_isht, (long long)FE, f1, sA1);
    L1D(k_pack_isht, (long long)FE, f2, sA2);
    gemm(stream, sA1, iwt, f3, nullptr, BC, NLAT, LMAX, MMAX,
         (long long)BC * LMAX, (long long)NLAT * LMAX, (long long)BC * NLAT, nullptr, 1.f, 0, 0, 0);
    gemm(stream, sA2, iwt, f4, nullptr, BC, NLAT, LMAX, MMAX,
         (long long)BC * LMAX, (long long)NLAT * LMAX, (long long)BC * NLAT, nullptr, 1.f, 0, 0, 0);

    // (5) irFFT as GEMM: [65536 x 288] x [256 x 288]^T
    L1D(k_pack_idft, (long long)BC * NLAT * K2PAD, f3, f4, abufA);
    gemm(stream, abufA, Gm, g1, nullptr, BC * NLAT, NLON, K2PAD, 1, 0, 0, 0, nullptr, 1.f, 0, 0, 0);

    // (6) gelu + residual
    L1D(k_gelu_add, (long long)HTOT, g1, hres, h, (long long)HTOT);

    // (7) MLP: h -> gelu(h W1 + b1) W2 + b2 + h
    L1D(k_pack_hT, (long long)HTOT, h, abufA);
    L1D(k_cvt_h, (long long)HIDDEN * EMBED, w1 + (long long)l * HIDDEN * EMBED, w1h,
        (long long)HIDDEN * EMBED);
    L1D(k_cvt_h, (long long)EMBED * HIDDEN, w2 + (long long)l * EMBED * HIDDEN, w2h,
        (long long)EMBED * HIDDEN);
    gemm(stream, abufA, w1h, nullptr, abufB, BATCH * PIX, HIDDEN, EMBED, 1, 0, 0, 0,
         b1 + (long long)l * HIDDEN, 1.f, 0, /*gelu*/1, /*f16 out*/1);
    gemm(stream, abufB, w2h, g1, nullptr, BATCH * PIX, EMBED, HIDDEN, 1, 0, 0, 0,
         b2 + (long long)l * EMBED, 1.f, 0, 0, 0);
    L1D(k_mlp_res, (long long)HTOT, g1, h);
  }

  // ---- decoder ----
  L1D(k_decoder, (long long)BATCH * COUT * PIX, h, w_dec, b_dec, out);
  (void)in_sizes; (void)n_in; (void)out_size; (void)ws_size;
}
